// EdgeNetwork_9096740732968
// MI455X (gfx1250) — compile-verified, matrix operand order checked by reading, not run on hardware
//
#include <hip/hip_runtime.h>

typedef __bf16 bf16x16 __attribute__((ext_vector_type(16)));
typedef float  f32x8   __attribute__((ext_vector_type(8)));
typedef float  f32x4   __attribute__((ext_vector_type(4)));

#define NT_TILES 68          // 64 weight col-tiles (16 cols each) + 4 bias col-tiles
#define KT_TILES 2           // K = 64 = 2 x 32
#define FRAG_ELEMS 512       // 32 lanes * 16 bf16 per B fragment
#define W2_ELEMS (KT_TILES * NT_TILES * FRAG_ELEMS)   // 69632 per matrix (hi / lo)
#define WAVES_PER_BLOCK 4

__global__ void zero_out_kernel(float* __restrict__ out, int n) {
  int i = blockIdx.x * blockDim.x + threadIdx.x;
  if (i < n) out[i] = 0.0f;
}

// Pre-pack W2[j, n] (n = k*64+i for n<1024, bias cols for n>=1024) into
// per-lane WMMA B-fragment order, split into bf16 hi/lo.
__global__ void build_w2_kernel(const float* __restrict__ kern, const float* __restrict__ bias,
                                __bf16* __restrict__ whi, __bf16* __restrict__ wlo) {
  int tid = blockIdx.x * blockDim.x + threadIdx.x;
  if (tid >= W2_ELEMS) return;
  int kt   = tid / (NT_TILES * FRAG_ELEMS);
  int rem  = tid % (NT_TILES * FRAG_ELEMS);
  int nt   = rem / FRAG_ELEMS;
  int s    = rem % FRAG_ELEMS;
  int lane = s >> 4;
  int e    = s & 15;                 // element index within lane fragment
  int half = lane >> 4;
  int c    = lane & 15;
  // K (reduction index j) for fragment element e, per 16-bit operand layout
  int K = kt * 32 + 8 * half + (e < 8 ? e : e + 8);
  int n = nt * 16 + c;
  float w;
  if (n < 1024) {
    int kk = n >> 6, i = n & 63;
    w = kern[kk * 4096 + i * 64 + K];     // kernel[k, i*64 + j]
  } else {
    int i = n - 1024;
    w = bias[i * 64 + K];                 // bias[i*64 + j]
  }
  __bf16 h = (__bf16)w;
  whi[tid] = h;
  wlo[tid] = (__bf16)(w - (float)h);
}

// One 16x16 D tile for column-tile nt: D = A * B2[nt] with 3xBF16 split.
__device__ __forceinline__ f32x8 wmma_tile(const bf16x16* a_hi, const bf16x16* a_lo,
                                           const __bf16* __restrict__ whi,
                                           const __bf16* __restrict__ wlo,
                                           int nt, int lane) {
  f32x8 d = {0.f, 0.f, 0.f, 0.f, 0.f, 0.f, 0.f, 0.f};
#pragma unroll
  for (int kt = 0; kt < KT_TILES; ++kt) {
    const long long fo = ((long long)(kt * NT_TILES + nt)) * FRAG_ELEMS + lane * 16;
    bf16x16 bh = *(const bf16x16*)(whi + fo);
    bf16x16 bl = *(const bf16x16*)(wlo + fo);
    // hi*hi + hi*lo + lo*hi  (~fp32 accuracy, drops only ~2^-16-relative lo*lo)
    d = __builtin_amdgcn_wmma_f32_16x16x32_bf16(false, a_hi[kt], false, bh, (short)0, d, false, false);
    d = __builtin_amdgcn_wmma_f32_16x16x32_bf16(false, a_hi[kt], false, bl, (short)0, d, false, false);
    d = __builtin_amdgcn_wmma_f32_16x16x32_bf16(false, a_lo[kt], false, bh, (short)0, d, false, false);
  }
  return d;
}

__global__ void __launch_bounds__(32 * WAVES_PER_BLOCK)
edge_network_fused(const float* __restrict__ atom, const float* __restrict__ bond,
                   const int* __restrict__ pair, const __bf16* __restrict__ whi,
                   const __bf16* __restrict__ wlo, float* __restrict__ out,
                   int n_edges, int n_tiles) {
  __shared__ float lds_bond[WAVES_PER_BLOCK][256];   // 16 edges x 16 bond feats per wave

  const int wv   = threadIdx.x >> 5;
  const int lane = threadIdx.x & 31;
  const int half = lane >> 4;
  const int c    = lane & 15;

  int tile = blockIdx.x * WAVES_PER_BLOCK + wv;
  const bool active = tile < n_tiles;
  if (!active) tile = n_tiles - 1;      // clamp: keep wave converged, skip stores later
  const int eb = tile * 16;

  // Stage this tile's bond features into LDS (contiguous 256 floats).
  {
    const long long glim  = (long long)n_edges * 16;
    const long long gbase = (long long)eb * 16 + lane * 8;
#pragma unroll
    for (int t = 0; t < 8; ++t) {
      long long g = gbase + t;
      lds_bond[wv][lane * 8 + t] = (g < glim) ? bond[g] : 0.0f;
    }
  }
  __syncthreads();

  // Gather nbr atom rows and build A fragments (f32 -> bf16 hi/lo split).
  bf16x16 a_hi[KT_TILES], a_lo[KT_TILES];
  {
    long long e = (long long)eb + c;
    if (e >= n_edges) e = n_edges - 1;
    const long long arow = (long long)pair[e * 2 + 1] * 64;
#pragma unroll
    for (int kt = 0; kt < KT_TILES; ++kt) {
      const int j0 = kt * 32 + 8 * half;
      f32x4 f0 = *(const f32x4*)(atom + arow + j0);
      f32x4 f1 = *(const f32x4*)(atom + arow + j0 + 4);
      f32x4 f2 = *(const f32x4*)(atom + arow + j0 + 16);
      f32x4 f3 = *(const f32x4*)(atom + arow + j0 + 20);
      float f[16];
#pragma unroll
      for (int i = 0; i < 4; ++i) { f[i] = f0[i]; f[4 + i] = f1[i]; f[8 + i] = f2[i]; f[12 + i] = f3[i]; }
#pragma unroll
      for (int i = 0; i < 16; ++i) {
        __bf16 h = (__bf16)f[i];
        a_hi[kt][i] = h;
        a_lo[kt][i] = (__bf16)(f[i] - (float)h);
      }
    }
  }

  float acc[8][4];
#pragma unroll
  for (int r = 0; r < 8; ++r)
#pragma unroll
    for (int ib = 0; ib < 4; ++ib) acc[r][ib] = 0.0f;

  // Main loop over the 16 bond coefficients; the 4 column-subtiles per
  // coefficient are fully unrolled so acc[][] indices are compile-time
  // constants (no v_movrel dynamic-VGPR indexing).
  for (int kk = 0; kk < 16; ++kk) {
    float bvals[8];
#pragma unroll
    for (int r = 0; r < 8; ++r)
      bvals[r] = lds_bond[wv][(r + 8 * half) * 16 + kk];
#pragma unroll
    for (int ib = 0; ib < 4; ++ib) {
      f32x8 d = wmma_tile(a_hi, a_lo, whi, wlo, kk * 4 + ib, lane);
#pragma unroll
      for (int r = 0; r < 8; ++r)
        acc[r][ib] += bvals[r] * d[r];
    }
  }

  // Bias column-tiles (multiplier 1), constant ib.
#pragma unroll
  for (int ib = 0; ib < 4; ++ib) {
    f32x8 d = wmma_tile(a_hi, a_lo, whi, wlo, 64 + ib, lane);
#pragma unroll
    for (int r = 0; r < 8; ++r)
      acc[r][ib] += d[r];
  }

  // Segment-sum: scatter-add msg rows into agg[tgt].
#pragma unroll
  for (int r = 0; r < 8; ++r) {
    const long long e = (long long)eb + r + 8 * half;
    if (active && e < n_edges) {
      const long long obase = (long long)pair[e * 2] * 64 + c;
#pragma unroll
      for (int ib = 0; ib < 4; ++ib)
        __hip_atomic_fetch_add(out + obase + ib * 16, acc[r][ib],
                               __ATOMIC_RELAXED, __HIP_MEMORY_SCOPE_AGENT);
    }
  }
}

extern "C" void kernel_launch(void* const* d_in, const int* in_sizes, int n_in,
                              void* d_out, int out_size, void* d_ws, size_t ws_size,
                              hipStream_t stream) {
  const float* atom = (const float*)d_in[0];   // (n_atoms, 64) f32
  const float* bond = (const float*)d_in[1];   // (n_edges, 16) f32
  const int*   pair = (const int*)d_in[2];     // (n_edges, 2) int
  const float* kern = (const float*)d_in[3];   // (16, 4096) f32
  const float* bias = (const float*)d_in[4];   // (4096,) f32
  float* out = (float*)d_out;

  const int n_edges = in_sizes[1] / 16;
  const int n_tiles = (n_edges + 15) / 16;

  __bf16* whi = (__bf16*)d_ws;                 // needs 2 * 69632 * 2 B = 272 KB of ws
  __bf16* wlo = whi + W2_ELEMS;

  zero_out_kernel<<<(out_size + 255) / 256, 256, 0, stream>>>(out, out_size);
  build_w2_kernel<<<(W2_ELEMS + 255) / 256, 256, 0, stream>>>(kern, bias, whi, wlo);

  const int nblocks = (n_tiles + WAVES_PER_BLOCK - 1) / WAVES_PER_BLOCK;
  edge_network_fused<<<nblocks, 32 * WAVES_PER_BLOCK, 0, stream>>>(
      atom, bond, pair, whi, wlo, out, n_edges, n_tiles);
}
